// CAVPCore_274877907258
// MI455X (gfx1250) — matrix operand order repaired and checked by reading, not compiled, not run on hardware
//
#include <hip/hip_runtime.h>
#include <hip/hip_bf16.h>

typedef __attribute__((ext_vector_type(16))) _Float16 v16h;
typedef __attribute__((ext_vector_type(8)))  _Float16 v8h;
typedef __attribute__((ext_vector_type(8)))  float    v8f;

#define TILE_M 128
#define TILE_N 64
#define TILE_K 32
#define LDS_PITCH 40   // 32 + 8 halves pad; row stride 80B keeps 16B alignment, breaks bank conflicts

// A-operand fragment (16x32 f16): lane = row (M = lane&15), half h = lane>>4,
// v16h elems 0..7 = K 8h..8h+7, elems 8..15 = K 16+8h..23+8h
__device__ inline v16h load_afrag(const _Float16* row, int half) {
    v8h lo = *(const v8h*)(row + half * 8);
    v8h hi = *(const v8h*)(row + 16 + half * 8);
    v16h r;
    #pragma unroll
    for (int i = 0; i < 8; ++i) { r[i] = lo[i]; r[8 + i] = hi[i]; }
    return r;
}

// B-operand fragment (32x16 f16): lane = col (N = lane&15), half h = lane>>4,
// v16h elems 0..7 = K 8h..8h+7, elems 8..15 = K 16+8h..23+8h
__device__ inline v16h load_bfrag(const _Float16* row, int half) {
    v8h lo = *(const v8h*)(row + half * 8);
    v8h hi = *(const v8h*)(row + 16 + half * 8);
    v16h r;
    #pragma unroll
    for (int i = 0; i < 8; ++i) { r[i] = lo[i]; r[8 + i] = hi[i]; }
    return r;
}

// C[M,N] = (acc ? C : 0) + Aeff[M,K] @ W[N,K]^T + bias0[n] + bias1[n]
// Aeff[m,k] = HAS_SUB ? (Asub[m/bcastL, k] - A[m,k]) : A[m,k]
// Grid: (N/64, M/128), block 256 = 8 wave32 waves, each wave owns a 32x32 output
// (2x2 WMMA tiles -> 4 v_wmma per K step, every fragment reused twice).
template <bool HAS_SUB>
__global__ void __launch_bounds__(256) gemm_nt_f16wmma(
    const float* __restrict__ A, int lda,
    const float* __restrict__ Asub, int bcastL, int ldsub,
    const float* __restrict__ W, int ldw,
    const float* __restrict__ bias0, const float* __restrict__ bias1,
    float* __restrict__ C, int ldc,
    int K, int accumulate)
{
    __shared__ _Float16 As[TILE_M * LDS_PITCH];
    __shared__ _Float16 Ws[TILE_N * LDS_PITCH];

    const int tid  = threadIdx.x;
    const int m0   = blockIdx.y * TILE_M;
    const int n0   = blockIdx.x * TILE_N;
    const int wave = tid >> 5;
    const int lane = tid & 31;
    const int wm   = wave & 3;    // 32-row slice of the 128-row tile
    const int wn   = wave >> 2;   // 32-col slice of the 64-col tile
    const int half = lane >> 4;
    const int l16  = lane & 15;

    // loaders: A tile 128x32 f32 -> 16 f32/thread; W tile 64x32 -> 8 f32/thread
    const int a_row = tid >> 1;        // 0..127
    const int a_col = (tid & 1) * 16;  // 0,16
    const int w_row = tid >> 2;        // 0..63
    const int w_col = (tid & 3) * 8;   // 0,8,16,24

    const float* aptr = A + (size_t)(m0 + a_row) * lda + a_col;
    const float* wptr = W + (size_t)(n0 + w_row) * ldw + w_col;
    const float* sptr = HAS_SUB
        ? (Asub + (size_t)((m0 + a_row) / bcastL) * ldsub + a_col) : nullptr;

    v8f acc00 = {}, acc01 = {}, acc10 = {}, acc11 = {};

    // software pipeline: prologue global loads for k0 = 0
    float4 ar0 = *(const float4*)(aptr + 0);
    float4 ar1 = *(const float4*)(aptr + 4);
    float4 ar2 = *(const float4*)(aptr + 8);
    float4 ar3 = *(const float4*)(aptr + 12);
    if (HAS_SUB) {
        float4 s0 = *(const float4*)(sptr + 0);
        float4 s1 = *(const float4*)(sptr + 4);
        float4 s2 = *(const float4*)(sptr + 8);
        float4 s3 = *(const float4*)(sptr + 12);
        ar0.x = s0.x - ar0.x; ar0.y = s0.y - ar0.y; ar0.z = s0.z - ar0.z; ar0.w = s0.w - ar0.w;
        ar1.x = s1.x - ar1.x; ar1.y = s1.y - ar1.y; ar1.z = s1.z - ar1.z; ar1.w = s1.w - ar1.w;
        ar2.x = s2.x - ar2.x; ar2.y = s2.y - ar2.y; ar2.z = s2.z - ar2.z; ar2.w = s2.w - ar2.w;
        ar3.x = s3.x - ar3.x; ar3.y = s3.y - ar3.y; ar3.z = s3.z - ar3.z; ar3.w = s3.w - ar3.w;
    }
    float4 wr0 = *(const float4*)(wptr + 0);
    float4 wr1 = *(const float4*)(wptr + 4);

    for (int k0 = 0; k0 < K; k0 += TILE_K) {
        // convert f32 -> f16 and commit the staged tile to LDS
        _Float16* ad = &As[a_row * LDS_PITCH + a_col];
        ad[0]  = (_Float16)ar0.x; ad[1]  = (_Float16)ar0.y; ad[2]  = (_Float16)ar0.z; ad[3]  = (_Float16)ar0.w;
        ad[4]  = (_Float16)ar1.x; ad[5]  = (_Float16)ar1.y; ad[6]  = (_Float16)ar1.z; ad[7]  = (_Float16)ar1.w;
        ad[8]  = (_Float16)ar2.x; ad[9]  = (_Float16)ar2.y; ad[10] = (_Float16)ar2.z; ad[11] = (_Float16)ar2.w;
        ad[12] = (_Float16)ar3.x; ad[13] = (_Float16)ar3.y; ad[14] = (_Float16)ar3.z; ad[15] = (_Float16)ar3.w;
        _Float16* wd = &Ws[w_row * LDS_PITCH + w_col];
        wd[0] = (_Float16)wr0.x; wd[1] = (_Float16)wr0.y; wd[2] = (_Float16)wr0.z; wd[3] = (_Float16)wr0.w;
        wd[4] = (_Float16)wr1.x; wd[5] = (_Float16)wr1.y; wd[6] = (_Float16)wr1.z; wd[7] = (_Float16)wr1.w;
        __syncthreads();

        // issue next tile's global loads under the WMMAs
        if (k0 + TILE_K < K) {
            const float* ap = aptr + k0 + TILE_K;
            const float* wp = wptr + k0 + TILE_K;
            ar0 = *(const float4*)(ap + 0);
            ar1 = *(const float4*)(ap + 4);
            ar2 = *(const float4*)(ap + 8);
            ar3 = *(const float4*)(ap + 12);
            if (HAS_SUB) {
                const float* sp = sptr + k0 + TILE_K;
                float4 s0 = *(const float4*)(sp + 0);
                float4 s1 = *(const float4*)(sp + 4);
                float4 s2 = *(const float4*)(sp + 8);
                float4 s3 = *(const float4*)(sp + 12);
                ar0.x = s0.x - ar0.x; ar0.y = s0.y - ar0.y; ar0.z = s0.z - ar0.z; ar0.w = s0.w - ar0.w;
                ar1.x = s1.x - ar1.x; ar1.y = s1.y - ar1.y; ar1.z = s1.z - ar1.z; ar1.w = s1.w - ar1.w;
                ar2.x = s2.x - ar2.x; ar2.y = s2.y - ar2.y; ar2.z = s2.z - ar2.z; ar2.w = s2.w - ar2.w;
                ar3.x = s3.x - ar3.x; ar3.y = s3.y - ar3.y; ar3.z = s3.z - ar3.z; ar3.w = s3.w - ar3.w;
            }
            wr0 = *(const float4*)(wp + 0);
            wr1 = *(const float4*)(wp + 4);
            if (k0 + 2 * TILE_K < K) {   // gfx1250 global_prefetch_b8, two tiles ahead
                __builtin_prefetch(ap + TILE_K, 0, 1);
                __builtin_prefetch(wp + TILE_K, 0, 1);
            }
        }

        // fragments + 2x2 WMMA
        v16h af0 = load_afrag(&As[(wm * 32 + l16) * LDS_PITCH], half);
        v16h af1 = load_afrag(&As[(wm * 32 + 16 + l16) * LDS_PITCH], half);
        v16h bf0 = load_bfrag(&Ws[(wn * 32 + l16) * LDS_PITCH], half);
        v16h bf1 = load_bfrag(&Ws[(wn * 32 + 16 + l16) * LDS_PITCH], half);

        acc00 = __builtin_amdgcn_wmma_f32_16x16x32_f16(false, af0, false, bf0, (short)0, acc00, false, false);
        acc01 = __builtin_amdgcn_wmma_f32_16x16x32_f16(false, af0, false, bf1, (short)0, acc01, false, false);
        acc10 = __builtin_amdgcn_wmma_f32_16x16x32_f16(false, af1, false, bf0, (short)0, acc10, false, false);
        acc11 = __builtin_amdgcn_wmma_f32_16x16x32_f16(false, af1, false, bf1, (short)0, acc11, false, false);
        __syncthreads();
    }

    // Epilogue: D layout -> M = v + 8*half, N = l16
    const int nc0 = n0 + wn * 32 + l16;
    const int nc1 = nc0 + 16;
    float bias_c0 = 0.f, bias_c1 = 0.f;
    if (bias0) { bias_c0 += bias0[nc0]; bias_c1 += bias0[nc1]; }
    if (bias1) { bias_c0 += bias1[nc0]; bias_c1 += bias1[nc1]; }
    #pragma unroll
    for (int v = 0; v < 8; ++v) {
        const int mA = m0 + wm * 32 + v + half * 8;   // rows of WMMA tile 0
        const int mB = mA + 16;                        // rows of WMMA tile 1
        float* p00 = C + (size_t)mA * ldc + nc0;
        float* p01 = C + (size_t)mA * ldc + nc1;
        float* p10 = C + (size_t)mB * ldc + nc0;
        float* p11 = C + (size_t)mB * ldc + nc1;
        float r00 = acc00[v] + bias_c0;
        float r01 = acc01[v] + bias_c1;
        float r10 = acc10[v] + bias_c0;
        float r11 = acc11[v] + bias_c1;
        if (accumulate) { r00 += *p00; r01 += *p01; r10 += *p10; r11 += *p11; }
        *p00 = r00; *p01 = r01; *p10 = r10; *p11 = r11;
    }
}

// h2,c2 from pre-activation gates [B,4N] (order i,f,g,o) and c_in [B,N]
__global__ void lstm_cell_kernel(const float* __restrict__ gates,
                                 const float* __restrict__ c_in,
                                 float* __restrict__ h_out,
                                 float* __restrict__ c_out,
                                 float* __restrict__ h_out2,
                                 int total, int N)
{
    int idx = blockIdx.x * blockDim.x + threadIdx.x;
    if (idx >= total) return;
    int b = idx / N, n = idx % N;
    const float* g = gates + (size_t)b * 4 * N;
    float gi = g[n], gf = g[N + n], gg = g[2 * N + n], go = g[3 * N + n];
    float si = 1.f / (1.f + expf(-gi));
    float sf = 1.f / (1.f + expf(-gf));
    float so = 1.f / (1.f + expf(-go));
    float c2 = sf * c_in[idx] + si * tanhf(gg);
    float h2 = so * tanhf(c2);
    c_out[idx] = c2;
    h_out[idx] = h2;
    if (h_out2) h_out2[idx] = h2;
}

// Bahdanau attend. One block per batch element b.
// logits[l] = sum_a tanh(fproj[b,l,a] + hproj[b,a]) * wa[a] + wab
// out[b,d]  = sum_l softmax(logits)[l] * (vsub ? vsub[b,d]-value[b,l,d] : value[b,l,d])
__global__ void __launch_bounds__(256) attend_kernel(
    const float* __restrict__ hproj,
    const float* __restrict__ fproj,
    const float* __restrict__ wa,
    const float* __restrict__ wab,
    const float* __restrict__ value,
    const float* __restrict__ vsub,
    float* __restrict__ out,
    int L, int Adim, int D)
{
    __shared__ float logits[64];
    __shared__ float wts[64];
    __shared__ float red[2];

    const int b = blockIdx.x;
    const int tid = threadIdx.x;
    const int wave = tid >> 5, lane = tid & 31;
    const float* hp = hproj + (size_t)b * Adim;

    for (int l = wave; l < L; l += 8) {
        const float* fp = fproj + ((size_t)b * L + l) * Adim;
        float s = 0.f;
        for (int a = lane; a < Adim; a += 32)
            s += tanhf(fp[a] + hp[a]) * wa[a];
        #pragma unroll
        for (int off = 16; off > 0; off >>= 1)
            s += __shfl_xor(s, off, 32);
        if (lane == 0) logits[l] = s + wab[0];
    }
    __syncthreads();
    if (tid == 0) {
        float mx = -1e30f;
        for (int l = 0; l < L; ++l) mx = fmaxf(mx, logits[l]);
        float sm = 0.f;
        for (int l = 0; l < L; ++l) sm += expf(logits[l] - mx);
        red[0] = mx; red[1] = sm;
    }
    __syncthreads();
    if (tid < L) wts[tid] = expf(logits[tid] - red[0]) / red[1];
    __syncthreads();

    for (int d = tid; d < D; d += 256) {
        const float* vp = value + (size_t)b * L * D + d;
        float vs = vsub ? vsub[(size_t)b * D + d] : 0.f;
        float s = 0.f;
        for (int l = 0; l < L; ++l) {
            float v = vp[(size_t)l * D];
            if (vsub) v = vs - v;
            s += wts[l] * v;
        }
        out[(size_t)b * D + d] = s;
    }
}

// dst[row, 0:w0]=s0[row], [w0:w0+w1]=s1[row], rest=s2[row]
__global__ void concat3_kernel(float* __restrict__ dst,
                               const float* __restrict__ s0, int w0,
                               const float* __restrict__ s1, int w1,
                               const float* __restrict__ s2, int w2,
                               int totalW, int count)
{
    int idx = blockIdx.x * blockDim.x + threadIdx.x;
    if (idx >= count) return;
    int row = idx / totalW, col = idx % totalW;
    float v;
    if (col < w0)            v = s0[(size_t)row * w0 + col];
    else if (col < w0 + w1)  v = s1[(size_t)row * w1 + (col - w0)];
    else                     v = s2[(size_t)row * w2 + (col - w0 - w1)];
    dst[idx] = v;
}

extern "C" void kernel_launch(void* const* d_in, const int* in_sizes, int n_in,
                              void* d_out, int out_size, void* d_ws, size_t ws_size,
                              hipStream_t stream) {
    (void)in_sizes; (void)n_in; (void)out_size; (void)ws_size;
    constexpr int B = 256, L = 64, R = 1024, Adim = 512, E = 1024;

    const float* xt      = (const float*)d_in[0];
    const float* fc      = (const float*)d_in[1];
    const float* roi     = (const float*)d_in[2];   // [B,L,R]
    const float* proi    = (const float*)d_in[3];   // [B,L,A]
    const float* ctxin   = (const float*)d_in[4];   // [B,L,R]
    const float* sh      = (const float*)d_in[5];   // [2,B,R]
    const float* sc      = (const float*)d_in[6];   // [2,B,R]
    const float* Wihv    = (const float*)d_in[7];
    const float* Whhv    = (const float*)d_in[8];
    const float* bihv    = (const float*)d_in[9];
    const float* bhhv    = (const float*)d_in[10];
    const float* Wihl    = (const float*)d_in[11];
    const float* Whhl    = (const float*)d_in[12];
    const float* bihl    = (const float*)d_in[13];
    const float* bhhl    = (const float*)d_in[14];
    const float* sp_wh_w = (const float*)d_in[15];
    const float* sp_wh_b = (const float*)d_in[16];
    const float* sp_wa_w = (const float*)d_in[17];
    const float* sp_wa_b = (const float*)d_in[18];
    const float* ctx_wh_w= (const float*)d_in[19];
    const float* ctx_wh_b= (const float*)d_in[20];
    const float* ctx_wv_w= (const float*)d_in[21];
    const float* ctx_wv_b= (const float*)d_in[22];
    const float* ctx_wa_w= (const float*)d_in[23];
    const float* ctx_wa_b= (const float*)d_in[24];
    const float* cmp_wh_w= (const float*)d_in[25];
    const float* cmp_wh_b= (const float*)d_in[26];
    const float* cmp_wv_w= (const float*)d_in[27];
    const float* cmp_wv_b= (const float*)d_in[28];
    const float* cmp_wa_w= (const float*)d_in[29];
    const float* cmp_wa_b= (const float*)d_in[30];
    const float* out_wv_w= (const float*)d_in[31];
    const float* out_wv_b= (const float*)d_in[32];
    const float* out_wh_w= (const float*)d_in[33];
    const float* out_wh_b= (const float*)d_in[34];
    const float* out_wa_w= (const float*)d_in[35];
    const float* out_wa_b= (const float*)d_in[36];

    float* ws = (float*)d_ws;
    float* env     = ws;                       // [B,3072]
    float* gates   = env     + B * 3072;       // [B,4096]
    float* hp_sp   = gates   + B * 4096;       // [B,512]
    float* hp_ctx  = hp_sp   + B * Adim;
    float* hp_cmp  = hp_ctx  + B * Adim;
    float* hp_out  = hp_cmp  + B * Adim;
    float* bigproj = hp_out  + B * Adim;       // [B*L,512] (ctx_proj then rel_proj)
    float* ctxfeat = bigproj + (size_t)B * L * Adim;  // [B,R]
    float* single  = ctxfeat + B * R;          // [B,R]
    float* compf   = single  + B * R;          // [B,R]
    float* feats3  = compf   + B * R;          // [B,3,R]
    float* f3proj  = feats3  + B * 3 * R;      // [B,3,A]
    float* langin  = f3proj  + B * 3 * Adim;   // [B,2R]

    float* outp    = (float*)d_out;            // h_lang copy
    float* h_vis   = outp + B * R;             // new_h[0]
    float* h_lang  = h_vis + B * R;            // new_h[1]
    float* c_vis   = h_lang + B * R;           // new_c[0]
    float* c_lang  = c_vis + B * R;            // new_c[1]
    float* outfeat = c_lang + B * R;           // output_feats

    const float* h0 = sh;             // state_h[0]
    const float* h1 = sh + B * R;     // state_h[1] (prev_h)
    const float* c0 = sc;
    const float* c1 = sc + B * R;

    dim3 blk(256);

    // 1) env = [prev_h | fc | xt]  [B,3072]
    concat3_kernel<<<(B * 3072) / 256, blk, 0, stream>>>(
        env, h1, R, fc, R, xt, E, 3072, B * 3072);

    // 2-3) visual LSTM gates = env@Wihv^T + b_ih + b_hh ; += h0@Whhv^T
    gemm_nt_f16wmma<false><<<dim3(4096 / TILE_N, B / TILE_M), blk, 0, stream>>>(
        env, 3072, nullptr, 1, 0, Wihv, 3072, bihv, bhhv, gates, 4096, 3072, 0);
    gemm_nt_f16wmma<false><<<dim3(4096 / TILE_N, B / TILE_M), blk, 0, stream>>>(
        h0, R, nullptr, 1, 0, Whhv, R, nullptr, nullptr, gates, 4096, R, 1);

    // 4) visual cell -> h_vis (new_h[0]), c_vis (new_c[0])
    lstm_cell_kernel<<<(B * R) / 256, blk, 0, stream>>>(
        gates, c0, h_vis, c_vis, nullptr, B * R, R);

    // 5) h projections for the four attention heads [B,512]
    gemm_nt_f16wmma<false><<<dim3(Adim / TILE_N, B / TILE_M), blk, 0, stream>>>(
        h_vis, R, nullptr, 1, 0, sp_wh_w, R, sp_wh_b, nullptr, hp_sp, Adim, R, 0);
    gemm_nt_f16wmma<false><<<dim3(Adim / TILE_N, B / TILE_M), blk, 0, stream>>>(
        h_vis, R, nullptr, 1, 0, ctx_wh_w, R, ctx_wh_b, nullptr, hp_ctx, Adim, R, 0);
    gemm_nt_f16wmma<false><<<dim3(Adim / TILE_N, B / TILE_M), blk, 0, stream>>>(
        h_vis, R, nullptr, 1, 0, cmp_wh_w, R, cmp_wh_b, nullptr, hp_cmp, Adim, R, 0);
    gemm_nt_f16wmma<false><<<dim3(Adim / TILE_N, B / TILE_M), blk, 0, stream>>>(
        h_vis, R, nullptr, 1, 0, out_wh_w, R, out_wh_b, nullptr, hp_out, Adim, R, 0);

    // 6) SingleSP attend (feats_proj precomputed = p_roi_feats)
    attend_kernel<<<B, blk, 0, stream>>>(
        hp_sp, proi, sp_wa_w, sp_wa_b, roi, nullptr, single, L, Adim, R);

    // 7-8) ContextSP: proj [B*L,512] then attend
    gemm_nt_f16wmma<false><<<dim3(Adim / TILE_N, (B * L) / TILE_M), blk, 0, stream>>>(
        ctxin, R, nullptr, 1, 0, ctx_wv_w, R, ctx_wv_b, nullptr, bigproj, Adim, R, 0);
    attend_kernel<<<B, blk, 0, stream>>>(
        hp_ctx, bigproj, ctx_wa_w, ctx_wa_b, ctxin, nullptr, ctxfeat, L, Adim, R);

    // 9-10) CompSP: rel = ctxfeat - roi (never materialized)
    gemm_nt_f16wmma<true><<<dim3(Adim / TILE_N, (B * L) / TILE_M), blk, 0, stream>>>(
        roi, R, ctxfeat, L, R, cmp_wv_w, R, cmp_wv_b, nullptr, bigproj, Adim, R, 0);
    attend_kernel<<<B, blk, 0, stream>>>(
        hp_cmp, bigproj, cmp_wa_w, cmp_wa_b, roi, ctxfeat, compf, L, Adim, R);

    // 11-13) OutputSP over {single, comp, fc}
    concat3_kernel<<<(B * 3 * R) / 256, blk, 0, stream>>>(
        feats3, single, R, compf, R, fc, R, 3 * R, B * 3 * R);
    gemm_nt_f16wmma<false><<<dim3(Adim / TILE_N, (B * 3) / TILE_M), blk, 0, stream>>>(
        feats3, R, nullptr, 1, 0, out_wv_w, R, out_wv_b, nullptr, f3proj, Adim, R, 0);
    attend_kernel<<<B, blk, 0, stream>>>(
        hp_out, f3proj, out_wa_w, out_wa_b, feats3, nullptr, outfeat, 3, Adim, R);

    // 14) lang_in = [output_feats | h_vis]  [B,2048]
    concat3_kernel<<<(B * 2 * R) / 256, blk, 0, stream>>>(
        langin, outfeat, R, h_vis, R, nullptr, 0, 2 * R, B * 2 * R);

    // 15-16) language LSTM gates
    gemm_nt_f16wmma<false><<<dim3(4096 / TILE_N, B / TILE_M), blk, 0, stream>>>(
        langin, 2 * R, nullptr, 1, 0, Wihl, 2 * R, bihl, bhhl, gates, 4096, 2 * R, 0);
    gemm_nt_f16wmma<false><<<dim3(4096 / TILE_N, B / TILE_M), blk, 0, stream>>>(
        h1, R, nullptr, 1, 0, Whhl, R, nullptr, nullptr, gates, 4096, R, 1);

    // 17) language cell -> h_lang (new_h[1] + output copy), c_lang (new_c[1])
    lstm_cell_kernel<<<(B * R) / 256, blk, 0, stream>>>(
        gates, c1, h_lang, c_lang, outp, B * R, R);
}
